// CCL_Module_78082505441363
// MI455X (gfx1250) — matrix operand order repaired and checked by reading, not compiled, not run on hardware
//
#include <hip/hip_runtime.h>

typedef __attribute__((ext_vector_type(16))) _Float16 v16h;
typedef __attribute__((ext_vector_type(8)))  _Float16 v8h;
typedef __attribute__((ext_vector_type(8)))  float    v8f;

constexpr int B = 4, C = 32, H = 64, W = 64;
constexpr int L = H * W;          // 4096
constexpr int D = C * 9;          // 288 = 9 chunks of K=32
constexpr int TR = 32;            // l-tile rows per iteration
constexpr int NT = L / TR;        // 128 iterations
constexpr int TILE_BYTES = TR * D * 2;   // 18432 B per buffer
constexpr float SM_SCALE = 10.0f;

// ---------------------------------------------------------------------------
// Kernel 1: channel-wise L2 normalization, fp32 -> f16
// ---------------------------------------------------------------------------
__global__ void norm_kernel(const float* __restrict__ f1,
                            const float* __restrict__ f2,
                            _Float16* __restrict__ N1,
                            _Float16* __restrict__ N2) {
  int idx = blockIdx.x * blockDim.x + threadIdx.x;     // B*L threads
  if (idx >= B * L) return;
  int b = idx >> 12;
  int p = idx & (L - 1);
  const float* p1 = f1 + (size_t)b * C * L + p;
  const float* p2 = f2 + (size_t)b * C * L + p;
  float s1 = 0.f, s2 = 0.f;
  #pragma unroll
  for (int c = 0; c < C; ++c) {
    float a = p1[c * L]; s1 += a * a;
    float d = p2[c * L]; s2 += d * d;
  }
  float i1 = 1.0f / fmaxf(sqrtf(s1), 1e-12f);
  float i2 = 1.0f / fmaxf(sqrtf(s2), 1e-12f);
  _Float16* o1 = N1 + (size_t)b * C * L + p;
  _Float16* o2 = N2 + (size_t)b * C * L + p;
  #pragma unroll
  for (int c = 0; c < C; ++c) {
    o1[c * L] = (_Float16)(p1[c * L] * i1);
    o2[c * L] = (_Float16)(p2[c * L] * i2);
  }
}

// ---------------------------------------------------------------------------
// Kernel 2: 3x3 patch extraction into row-major [L, 288] f16 matrices
// ---------------------------------------------------------------------------
__global__ void patch_kernel(const _Float16* __restrict__ N1,
                             const _Float16* __restrict__ N2,
                             _Float16* __restrict__ Qm,
                             _Float16* __restrict__ Km) {
  int tid = blockIdx.x * blockDim.x + threadIdx.x;     // C*B*L threads
  if (tid >= C * B * L) return;
  int l  = tid & (L - 1);
  int bc = tid >> 12;
  int b  = bc & (B - 1);
  int c  = bc >> 2;
  int y = l >> 6, x = l & 63;
  const _Float16* n1 = N1 + ((size_t)(b * C + c)) * L;
  const _Float16* n2 = N2 + ((size_t)(b * C + c)) * L;
  _Float16* q = Qm + ((size_t)(b * L + l)) * D + c * 9;
  _Float16* k = Km + ((size_t)(b * L + l)) * D + c * 9;
  #pragma unroll
  for (int dy = 0; dy < 3; ++dy) {
    #pragma unroll
    for (int dx = 0; dx < 3; ++dx) {
      int ys = y + dy - 1, xs = x + dx - 1;
      bool in = (ys >= 0) & (ys < H) & (xs >= 0) & (xs < W);
      _Float16 v1 = in ? n1[ys * W + xs] : (_Float16)0.0f;
      _Float16 v2 = in ? n2[ys * W + xs] : (_Float16)0.0f;
      q[dy * 3 + dx] = v1;
      k[dy * 3 + dx] = v2;
    }
  }
}

// ---------------------------------------------------------------------------
// Async staging helpers (CDNA5 async global->LDS, tracked by ASYNCcnt).
// Inline asm per CDNA5 ISA (portable across toolchains).
// ---------------------------------------------------------------------------
__device__ __forceinline__ void async_cp_b128(uint32_t lds_off, uint64_t gaddr) {
  asm volatile("global_load_async_to_lds_b128 %0, %1, off"
               :: "v"(lds_off), "v"(gaddr) : "memory");
}
__device__ __forceinline__ void wait_async0() {
  asm volatile("s_wait_asynccnt 0x0" ::: "memory");
}

// stage one 32x288 f16 tile (18432 B) cooperatively: thread t moves
// 4 x b128 at [t*64, t*64+64) plus (t<128) one more b128 at 16384 + t*16
__device__ __forceinline__ void stage_tile(const _Float16* gsrc,
                                           const _Float16* lbuf, int t) {
  uint64_t g = (uint64_t)(uintptr_t)gsrc;
  uint32_t l = (uint32_t)(uintptr_t)lbuf;   // low 32 bits = LDS byte offset
  int off = t * 64;
  #pragma unroll
  for (int j = 0; j < 4; ++j)
    async_cp_b128(l + off + j * 16, g + off + j * 16);
  if (t < 128) {
    int off2 = 16384 + t * 16;
    async_cp_b128(l + off2, g + off2);
  }
}

// ---------------------------------------------------------------------------
// Kernel 3: fused correlation GEMM (WMMA f16) + online softmax + flow.
// Block = 256 threads = 8 waves sharing double-buffered K l-tiles in LDS
// (async staged); each wave owns 16 query columns with the Q operand
// resident in 72 VGPRs.  Per iteration: 18 x v_wmma_f32_16x16x32_f16.
// ---------------------------------------------------------------------------
__global__ __launch_bounds__(256) void corr_flow_kernel(
    const _Float16* __restrict__ Qm,
    const _Float16* __restrict__ Km,
    float* __restrict__ out) {
  __shared__ __align__(16) _Float16 smem[2][TR * D];   // 2 x 18432 B

  const int t       = threadIdx.x;
  const int lane    = t & 31;
  const int wave    = t >> 5;
  const int b       = blockIdx.x >> 5;        // 32 blocks per batch
  const int blk     = blockIdx.x & 31;
  const int q_base  = blk * 128 + wave * 16;  // this wave's 16 queries
  const int halfsel = lane >> 4;              // 0: lanes 0-15, 1: lanes 16-31
  const int r       = lane & 15;

  // ---- resident B operand (Q columns): lanes 0-15 hold K=0..15,
  // lanes 16-31 hold K=16..31 of each 32-chunk
  const _Float16* qp = Qm + ((size_t)(b * L + q_base + r)) * D + halfsel * 16;
  v16h qreg[9];
  #pragma unroll
  for (int ck = 0; ck < 9; ++ck)
    qreg[ck] = *(const v16h*)(qp + ck * 32);

  const _Float16* kbase = Km + ((size_t)b * L) * D;

  float m = -__builtin_inff();
  float Z = 0.f, Sy = 0.f, Sx = 0.f;

  // prologue: stage tile 0 into buffer 0
  stage_tile(kbase, &smem[0][0], t);

  for (int i = 0; i < NT; ++i) {
    const int cur = i & 1;
    wait_async0();        // my async loads into buf[cur] complete
    __syncthreads();      // everyone's loads done; everyone done reading buf[1-cur]
    {                     // stage next tile into the other buffer
      int nt = (i + 1 < NT) ? (i + 1) : 0;
      stage_tile(kbase + (size_t)nt * TR * D, &smem[1 - cur][0], t);
    }

    // ---- compute from buf[cur]: A-operand fragments from LDS
    // (lane<16 -> d {0..15}, lane>=16 -> d {16..31} per 32-chunk)
    const _Float16* a0 = &smem[cur][0] + (size_t)r * D + halfsel * 16;
    const _Float16* a1 = a0 + 16 * D;
    v8f acc0 = {}, acc1 = {};
    #pragma unroll
    for (int ck = 0; ck < 9; ++ck) {
      v8h lo0 = *(const v8h*)(a0 + ck * 32);
      v8h hi0 = *(const v8h*)(a0 + ck * 32 + 8);
      v8h lo1 = *(const v8h*)(a1 + ck * 32);
      v8h hi1 = *(const v8h*)(a1 + ck * 32 + 8);
      v16h A0 = __builtin_shufflevector(lo0, hi0,
                0, 1, 2, 3, 4, 5, 6, 7, 8, 9, 10, 11, 12, 13, 14, 15);
      v16h A1 = __builtin_shufflevector(lo1, hi1,
                0, 1, 2, 3, 4, 5, 6, 7, 8, 9, 10, 11, 12, 13, 14, 15);
      acc0 = __builtin_amdgcn_wmma_f32_16x16x32_f16(
                 false, A0, false, qreg[ck], (short)0, acc0, false, false);
      acc1 = __builtin_amdgcn_wmma_f32_16x16x32_f16(
                 false, A1, false, qreg[ck], (short)0, acc1, false, false);
    }

    // ---- online softmax over the 32 l-rows of this tile (16 per lane)
    const int lb = i * TR;
    float s[16];
    float tm = -__builtin_inff();
    #pragma unroll
    for (int v = 0; v < 8; ++v) {
      s[v]     = acc0[v] * SM_SCALE;
      s[v + 8] = acc1[v] * SM_SCALE;
      tm = fmaxf(tm, fmaxf(s[v], s[v + 8]));
    }
    float nm    = fmaxf(m, tm);
    float scale = __expf(m - nm);          // exp(-inf)=0 on first tile
    Z *= scale; Sy *= scale; Sx *= scale;
    #pragma unroll
    for (int v = 0; v < 16; ++v) {
      float e   = __expf(s[v] - nm);
      int  lrow = lb + ((v >> 3) << 4) + halfsel * 8 + (v & 7);
      Z  += e;
      Sy += e * (float)(lrow >> 6);
      Sx += e * (float)(lrow & 63);
    }
    m = nm;
  }

  // ---- merge the two half-rows of each query column (lane ^ 16)
  float mo  = __shfl_xor(m, 16, 32);
  float Zo  = __shfl_xor(Z, 16, 32);
  float Syo = __shfl_xor(Sy, 16, 32);
  float Sxo = __shfl_xor(Sx, 16, 32);
  float M2 = fmaxf(m, mo);
  float sa = __expf(m - M2), sb = __expf(mo - M2);
  Z  = Z * sa + Zo * sb;
  Sy = Sy * sa + Syo * sb;
  Sx = Sx * sa + Sxo * sb;

  if (halfsel == 0) {
    int q = q_base + r;
    float qy = (float)(q >> 6), qx = (float)(q & 63);
    out[(size_t)b * 2 * L + q]     = Sx / Z - qx;   // flow_w (channel 0)
    out[(size_t)b * 2 * L + L + q] = Sy / Z - qy;   // flow_h (channel 1)
  }
}

// ---------------------------------------------------------------------------
extern "C" void kernel_launch(void* const* d_in, const int* in_sizes, int n_in,
                              void* d_out, int out_size, void* d_ws, size_t ws_size,
                              hipStream_t stream) {
  const float* f1 = (const float*)d_in[0];
  const float* f2 = (const float*)d_in[1];
  float* out = (float*)d_out;

  // workspace layout (halfs): N1 | N2 | Qm | Km  (~20.9 MB total)
  _Float16* N1 = (_Float16*)d_ws;
  _Float16* N2 = N1 + (size_t)B * C * L;
  _Float16* Qm = N2 + (size_t)B * C * L;
  _Float16* Km = Qm + (size_t)B * L * D;

  norm_kernel <<<(B * L + 255) / 256, 256, 0, stream>>>(f1, f2, N1, N2);
  patch_kernel<<<(B * L * C + 255) / 256, 256, 0, stream>>>(N1, N2, Qm, Km);
  corr_flow_kernel<<<B * 32, 256, 0, stream>>>(Qm, Km, out);
}
